// GeoAttention_80049600463710
// MI455X (gfx1250) — matrix operand order repaired and checked by reading, not compiled, not run on hardware
//
#include <hip/hip_runtime.h>
#include <hip/hip_bf16.h>
#include <math.h>

// ---------------------------------------------------------------------------
// GeoAttention (Bhattacharyya-geodesic attention) for gfx1250 / MI455X.
// All matmuls via v_wmma_f32_16x16x32_f16 (f16 in, f32 accumulate).
// Flash-style single pass over keys; key/value tiles staged into LDS with
// CDNA5 async global->LDS copies (ASYNCcnt), double-buffered per block.
// ---------------------------------------------------------------------------

#define BATCH 2
#define SEQ   2048
#define HEADS 8
#define HDIM  64
#define DIM   512
#define DIM2  1024
#define ROWS  (BATCH * SEQ)   // 4096

typedef __attribute__((ext_vector_type(16))) _Float16 v16h;
typedef __attribute__((ext_vector_type(8)))  _Float16 v8h;
typedef __attribute__((ext_vector_type(8)))  float    v8f;

// ---- WMMA fragment loaders (contiguous k order, same for A and B) -----
// A matrix 16x32 (MxK): lane holds row M=lane&15, K = half_id*16 .. half_id*16+15
// (contiguous, matching load_b_frag).  One contiguous 32-byte chunk per lane.
__device__ __forceinline__ v16h load_a_frag(const _Float16* row, int k0, int half_id) {
    return *(const v16h*)(row + k0 + half_id * 16);
}

// B matrix 32x16 (KxN): lane holds column N=lane&15; lanes 0-15 carry K=0..15,
// lanes 16-31 carry K=16..31 -> one contiguous 32-byte read from row N of B^T.
__device__ __forceinline__ v16h load_b_frag(const _Float16* bt_row, int k0, int khalf) {
    return *(const v16h*)(bt_row + k0 + khalf * 16);
}

// ---------------------------------------------------------------------------
// Down-convert kernels (one-time, tiny traffic)
// ---------------------------------------------------------------------------
__global__ void __launch_bounds__(256) cvt_xcat_kernel(
    const float* __restrict__ xr, const float* __restrict__ xi,
    _Float16* __restrict__ xcat) {
    int idx = blockIdx.x * blockDim.x + threadIdx.x;   // 0 .. ROWS*DIM2-1
    int c = idx & (DIM2 - 1);
    int r = idx >> 10;
    float v = (c < DIM) ? xr[(size_t)r * DIM + c] : xi[(size_t)r * DIM + (c - DIM)];
    xcat[idx] = (_Float16)v;
}

__global__ void __launch_bounds__(256) cvt_transpose_kernel(
    const float* __restrict__ src, _Float16* __restrict__ dst, int K, int N) {
    int idx = blockIdx.x * blockDim.x + threadIdx.x;
    if (idx >= K * N) return;
    int k = idx / N, n = idx - k * N;
    dst[(size_t)n * K + k] = (_Float16)src[idx];       // dst = src^T in f16
}

// ---------------------------------------------------------------------------
// Generic WMMA GEMM: C[M x N] = A[M x K](f16) * B[K x N](f16, given as B^T) + bias
// 1 wave -> one 16x16 output tile; block = 4 waves -> 64 rows.
// ---------------------------------------------------------------------------
__global__ void __launch_bounds__(128) gemm_f16_kernel(
    const _Float16* __restrict__ A, const _Float16* __restrict__ BT,
    const float* __restrict__ bias, float* __restrict__ C, int N, int K) {
    const int lane    = threadIdx.x & 31;
    const int wave    = threadIdx.x >> 5;
    const int half_id = lane >> 4;
    const int mn      = lane & 15;
    const int n0      = blockIdx.x * 16;
    const int i0      = blockIdx.y * 64 + wave * 16;

    const _Float16* arow = A  + (size_t)(i0 + mn) * K;
    const _Float16* brow = BT + (size_t)(n0 + mn) * K;

    v8f acc = {};
    for (int k0 = 0; k0 < K; k0 += 32) {
        __builtin_prefetch(arow + k0 + 256, 0, 3);   // WGP-scope prefetch
        __builtin_prefetch(brow + k0 + 256, 0, 3);
        v16h af = load_a_frag(arow, k0, half_id);
        v16h bf = load_b_frag(brow, k0, half_id);
        acc = __builtin_amdgcn_wmma_f32_16x16x32_f16(
            false, af, false, bf, (short)0, acc, false, false);
    }
    const float bv = bias[n0 + mn];
#pragma unroll
    for (int v = 0; v < 8; ++v) {
        int row = i0 + v + 8 * half_id;                // C/D layout: M = v + 8*(lane/16)
        C[(size_t)row * N + n0 + mn] = acc[v] + bv;
    }
}

// ---------------------------------------------------------------------------
// Row ops: mode 0/1 -> sp/sq = sqrt(softmax(row_64)) in f16 [B,H,S,64];
//          mode 2   -> V transposed per head to f16 [B,H,64,S].
// One wave per (b,h,s) row; lane owns d and d+32.
// ---------------------------------------------------------------------------
__global__ void __launch_bounds__(256) rowops_kernel(
    const float* __restrict__ Qf, const float* __restrict__ Kf, const float* __restrict__ Vf,
    _Float16* __restrict__ sp, _Float16* __restrict__ sq, _Float16* __restrict__ vt) {
    const int mode = blockIdx.y;
    const int lane = threadIdx.x & 31;
    const int wave = threadIdx.x >> 5;
    const int r = blockIdx.x * 8 + wave;               // 0 .. B*H*S-1
    const int s = r & (SEQ - 1);
    const int h = (r >> 11) & (HEADS - 1);
    const int b = r >> 14;
    const float* src = (mode == 0 ? Qf : (mode == 1 ? Kf : Vf))
                     + (size_t)(b * SEQ + s) * DIM + h * HDIM;
    float v0 = src[lane], v1 = src[lane + 32];
    if (mode < 2) {
        float m = fmaxf(v0, v1);
#pragma unroll
        for (int d = 16; d >= 1; d >>= 1) m = fmaxf(m, __shfl_xor(m, d, 32));
        float e0 = __expf(v0 - m), e1 = __expf(v1 - m);
        float sum = e0 + e1;
#pragma unroll
        for (int d = 16; d >= 1; d >>= 1) sum += __shfl_xor(sum, d, 32);
        float inv = 1.0f / sum;
        _Float16* dst = (mode == 0 ? sp : sq)
                      + (size_t)((b * HEADS + h) * SEQ + s) * HDIM;
        dst[lane]      = (_Float16)sqrtf(e0 * inv);
        dst[lane + 32] = (_Float16)sqrtf(e1 * inv);
    } else {
        _Float16* dst = vt + (size_t)(b * HEADS + h) * HDIM * SEQ + s;
        dst[(size_t)lane * SEQ]        = (_Float16)v0;
        dst[(size_t)(lane + 32) * SEQ] = (_Float16)v1;
    }
}

// ---------------------------------------------------------------------------
// Flash geodesic attention with CDNA5 async global->LDS staging.
//
// Block = 4 waves sharing one (b,h); each wave owns a 16-query tile.
// Per 32-key step the block stages the shared sq tile (32x64 f16, contiguous
// 4 KB) and vt tile (64 rows x 64 B) into LDS via GLOBAL_LOAD_ASYNC_TO_LDS_B128
// (GV mode: inst offset applies to both LDS and global address, ISA 08 section 4.4),
// double-buffered against the WMMA + acos/online-softmax compute.
//
// Dynamic LDS layout (no static LDS in this kernel, so segment starts at 0):
//   [    0 ..  8191]  sq tiles,  2 buffers x 4096 B  (32 rows x 64 halves)
//   [ 8192 .. 16383]  vt tiles,  2 buffers x 4096 B  (64 rows x 32 halves)
//   [16384 .. 20479]  p staging, 4 waves  x 1024 B  (16 rows x 32 halves)
// ---------------------------------------------------------------------------
#define ATTN_LDS_BYTES 20480

__device__ __forceinline__ void async_stage_tiles(
    const _Float16* sqb, const _Float16* vtb, int j0, int buf, int tid) {
    // sq tile: contiguous 4096 B; 128 lanes x 16 B x 2 instructions.
    {
        unsigned long long g = (unsigned long long)(const void*)(sqb + (size_t)j0 * HDIM)
                             + (unsigned)tid * 16u;
        unsigned int l = (unsigned)(buf * 4096 + tid * 16);
        asm volatile("global_load_async_to_lds_b128 %0, %1, off"
                     :: "v"(l), "v"(g) : "memory");
        asm volatile("global_load_async_to_lds_b128 %0, %1, off offset:2048"
                     :: "v"(l), "v"(g) : "memory");
    }
    // vt tile: 64 rows x 64 B (row stride SEQ halves). Lane -> (row, 16B chunk).
    {
        int r = tid >> 2;                    // 0..31
        int cb = (tid & 3) * 16;             // byte chunk within row
        unsigned long long g = (unsigned long long)(const void*)(vtb + (size_t)r * SEQ + j0)
                             + (unsigned)cb;
        unsigned int l = (unsigned)(8192 + buf * 4096 + r * 64 + cb);
        asm volatile("global_load_async_to_lds_b128 %0, %1, off"
                     :: "v"(l), "v"(g) : "memory");
        unsigned long long g2 = g + (unsigned long long)32 * SEQ * 2;  // rows 32..63
        unsigned int l2 = l + 2048u;
        asm volatile("global_load_async_to_lds_b128 %0, %1, off"
                     :: "v"(l2), "v"(g2) : "memory");
    }
}

__global__ void __launch_bounds__(128) attn_kernel(
    const _Float16* __restrict__ sp, const _Float16* __restrict__ sq,
    const _Float16* __restrict__ vt, _Float16* __restrict__ att) {
    extern __shared__ char smem[];
    const int tid     = threadIdx.x;
    const int lane    = tid & 31;
    const int wave    = tid >> 5;
    const int half_id = lane >> 4;
    const int n       = lane & 15;
    const int h  = blockIdx.y, b = blockIdx.z;
    const int i0 = blockIdx.x * 64 + wave * 16;

    const size_t bh = (size_t)(b * HEADS + h);
    const _Float16* spb = sp + bh * SEQ * HDIM;
    const _Float16* sqb = sq + bh * SEQ * HDIM;
    const _Float16* vtb = vt + bh * HDIM * SEQ;
    _Float16* pst = (_Float16*)(smem + 16384 + wave * 1024);   // [16][32]

    const _Float16* sprow = spb + (size_t)(i0 + n) * HDIM;
    v16h a0 = load_a_frag(sprow, 0,  half_id);         // K = 0..31  of head dim
    v16h a1 = load_a_frag(sprow, 32, half_id);         // K = 32..63

    v8f acc[4] = {};                                   // 16 rows x 64 head-dim cols
    float mrow[8], lrow[8];
#pragma unroll
    for (int v = 0; v < 8; ++v) { mrow[v] = -3.0e38f; lrow[v] = 0.0f; }

    // preload tile 0 into buffer 0
    async_stage_tiles(sqb, vtb, 0, 0, tid);

    const int NT = SEQ / 32;
    for (int jt = 0; jt < NT; ++jt) {
        const int j0  = jt * 32;
        const int buf = jt & 1;
        if (jt + 1 < NT) {
            async_stage_tiles(sqb, vtb, j0 + 32, buf ^ 1, tid);
            asm volatile("s_wait_asynccnt 0x4" ::: "memory");  // current tile landed
        } else {
            asm volatile("s_wait_asynccnt 0x0" ::: "memory");
        }
        __syncthreads();                                       // tile visible block-wide

        const _Float16* sqt = (const _Float16*)(smem + buf * 4096);          // [32][64]
        const _Float16* vtt = (const _Float16*)(smem + 8192 + buf * 4096);   // [64][32]

        float sv[16];
#pragma unroll
        for (int js = 0; js < 2; ++js) {
            const _Float16* sqrow = sqt + (size_t)(js * 16 + n) * HDIM;
            v16h bq0 = load_b_frag(sqrow, 0,  half_id);
            v16h bq1 = load_b_frag(sqrow, 32, half_id);
            v8f c = {};
            c = __builtin_amdgcn_wmma_f32_16x16x32_f16(false, a0, false, bq0, (short)0, c, false, false);
            c = __builtin_amdgcn_wmma_f32_16x16x32_f16(false, a1, false, bq1, (short)0, c, false, false);
#pragma unroll
            for (int v = 0; v < 8; ++v) {
                float bc = fminf(fmaxf(c[v], -1.0f), 1.0f);
                float d  = acosf(bc);
                sv[js * 8 + v] = -d * d;               // GAMMA = 1
            }
        }
        // Online-softmax update; rows live in the same (lane/16, v) slots as acc.
#pragma unroll
        for (int v = 0; v < 8; ++v) {
            float tm = fmaxf(sv[v], sv[8 + v]);
#pragma unroll
            for (int dd = 1; dd < 16; dd <<= 1) tm = fmaxf(tm, __shfl_xor(tm, dd, 32));
            float mnew  = fmaxf(mrow[v], tm);
            float scale = __expf(mrow[v] - mnew);
            float p0 = __expf(sv[v] - mnew);
            float p1 = __expf(sv[8 + v] - mnew);
            float ps = p0 + p1;
#pragma unroll
            for (int dd = 1; dd < 16; dd <<= 1) ps += __shfl_xor(ps, dd, 32);
            lrow[v] = lrow[v] * scale + ps;
            mrow[v] = mnew;
            acc[0][v] *= scale; acc[1][v] *= scale;
            acc[2][v] *= scale; acc[3][v] *= scale;
            pst[(v + 8 * half_id) * 32 + n]      = (_Float16)p0;
            pst[(v + 8 * half_id) * 32 + 16 + n] = (_Float16)p1;
        }
        asm volatile("s_wait_dscnt 0x0" ::: "memory");  // p tile visible to whole wave
        v16h pa = load_a_frag(pst + n * 32, 0, half_id);         // 16x32 A fragment
#pragma unroll
        for (int t = 0; t < 4; ++t) {
            v16h bv = load_b_frag(vtt + (size_t)(t * 16 + n) * 32, 0, half_id);
            acc[t] = __builtin_amdgcn_wmma_f32_16x16x32_f16(
                false, pa, false, bv, (short)0, acc[t], false, false);
        }
        __syncthreads();   // all waves done reading buf before it is re-staged
    }

    // Normalize and scatter to attended [B,S,DIM] (head-concat layout) in f16.
    _Float16* outb = att + (size_t)(b * SEQ + i0) * DIM + h * HDIM;
#pragma unroll
    for (int v = 0; v < 8; ++v) {
        int row = v + 8 * half_id;
        float inv = 1.0f / lrow[v];
        _Float16* orow = outb + (size_t)row * DIM;
#pragma unroll
        for (int t = 0; t < 4; ++t)
            orow[t * 16 + n] = (_Float16)(acc[t][v] * inv);
    }
}

// ---------------------------------------------------------------------------
extern "C" void kernel_launch(void* const* d_in, const int* in_sizes, int n_in,
                              void* d_out, int out_size, void* d_ws, size_t ws_size,
                              hipStream_t stream) {
    const float* x_real = (const float*)d_in[0];
    const float* x_imag = (const float*)d_in[1];
    const float* Wq = (const float*)d_in[2];  const float* bq = (const float*)d_in[3];
    const float* Wk = (const float*)d_in[4];  const float* bk = (const float*)d_in[5];
    const float* Wv = (const float*)d_in[6];  const float* bv = (const float*)d_in[7];
    const float* Wo = (const float*)d_in[8];  const float* bo = (const float*)d_in[9];
    float* out = (float*)d_out;

    // Workspace carving (256B aligned)
    char* base = (char*)d_ws;
    size_t off = 0;
    auto carve = [&](size_t bytes) -> char* {
        char* p = base + off;
        off = (off + bytes + 255) & ~(size_t)255;
        return p;
    };
    _Float16* xcat = (_Float16*)carve((size_t)ROWS * DIM2 * 2);    // [4096,1024] f16
    _Float16* WqT  = (_Float16*)carve((size_t)DIM  * DIM2 * 2);    // [512,1024]  f16
    _Float16* WkT  = (_Float16*)carve((size_t)DIM  * DIM2 * 2);
    _Float16* WvT  = (_Float16*)carve((size_t)DIM  * DIM2 * 2);
    _Float16* WoT  = (_Float16*)carve((size_t)DIM2 * DIM  * 2);    // [1024,512]  f16
    float*    Qf   = (float*)   carve((size_t)ROWS * DIM * 4);     // [4096,512]  f32
    float*    Kf   = (float*)   carve((size_t)ROWS * DIM * 4);
    float*    Vf   = (float*)   carve((size_t)ROWS * DIM * 4);
    _Float16* spb  = (_Float16*)carve((size_t)BATCH * HEADS * SEQ * HDIM * 2); // [B,H,S,64]
    _Float16* sqb  = (_Float16*)carve((size_t)BATCH * HEADS * SEQ * HDIM * 2);
    _Float16* vtb  = (_Float16*)carve((size_t)BATCH * HEADS * HDIM * SEQ * 2); // [B,H,64,S]
    _Float16* attb = (_Float16*)carve((size_t)ROWS * DIM * 2);     // attended f16

    // 1) down-convert inputs & transpose weights to f16
    cvt_xcat_kernel<<<(ROWS * DIM2) / 256, 256, 0, stream>>>(x_real, x_imag, xcat);
    cvt_transpose_kernel<<<(DIM2 * DIM + 255) / 256, 256, 0, stream>>>(Wq, WqT, DIM2, DIM);
    cvt_transpose_kernel<<<(DIM2 * DIM + 255) / 256, 256, 0, stream>>>(Wk, WkT, DIM2, DIM);
    cvt_transpose_kernel<<<(DIM2 * DIM + 255) / 256, 256, 0, stream>>>(Wv, WvT, DIM2, DIM);
    cvt_transpose_kernel<<<(DIM * DIM2 + 255) / 256, 256, 0, stream>>>(Wo, WoT, DIM, DIM2);

    // 2) QKV projections: [4096,1024] x [1024,512] via WMMA
    dim3 gP(DIM / 16, ROWS / 64);
    gemm_f16_kernel<<<gP, 128, 0, stream>>>(xcat, WqT, bq, Qf, DIM, DIM2);
    gemm_f16_kernel<<<gP, 128, 0, stream>>>(xcat, WkT, bk, Kf, DIM, DIM2);
    gemm_f16_kernel<<<gP, 128, 0, stream>>>(xcat, WvT, bv, Vf, DIM, DIM2);

    // 3) sqrt-softmax for Q,K; per-head transpose of V  (grid.y = mode)
    rowops_kernel<<<dim3((BATCH * HEADS * SEQ) / 8, 3), 256, 0, stream>>>(
        Qf, Kf, Vf, spb, sqb, vtb);

    // 4) flash geodesic attention (async LDS staging, double-buffered)
    attn_kernel<<<dim3(SEQ / 64, HEADS, BATCH), 128, ATTN_LDS_BYTES, stream>>>(
        spb, sqb, vtb, attb);

    // 5) output projection: [4096,512] x [512,1024] + bo -> d_out (f32)
    dim3 gO(DIM2 / 16, ROWS / 64);
    gemm_f16_kernel<<<gO, 128, 0, stream>>>(attb, WoT, bo, out, DIM2, DIM);
}
